// RNN_GRU_83631603187830
// MI455X (gfx1250) — matrix operand order repaired and checked by reading, not compiled, not run on hardware
//
#include <hip/hip_runtime.h>
#include <hip/hip_bf16.h>
#include <stddef.h>
#include <stdint.h>

// ---------------------------------------------------------------------------
// 3-layer zoneout-GRU stack for MI455X (gfx1250).
//  - All matmuls via v_wmma_f32_16x16x32_bf16 (wave32, 16x16 tiles).
//  - GEMM B-tiles staged in LDS with global_load_async_to_lds_b128
//    (ASYNCcnt-tracked, double-buffered).
//  - Recurrence: persistent kernel, 32 WGs, U-slices resident in LDS,
//    grid barrier via device-scope atomics, h state ping-pong in L2.
// ---------------------------------------------------------------------------

typedef __attribute__((ext_vector_type(16))) __bf16 v16bf;
typedef __attribute__((ext_vector_type(8)))  float  v8f;
typedef __attribute__((ext_vector_type(4)))  float  f32x4;

union FragBF { v16bf v; f32x4 q[2]; };

__device__ __forceinline__ __bf16 to_bf16(float x) {
    union { float f; unsigned u; } v; v.f = x;
    unsigned r = (v.u + 0x7FFFu + ((v.u >> 16) & 1u)) >> 16;   // RNE
    unsigned short h = (unsigned short)r;
    return __builtin_bit_cast(__bf16, h);
}
__device__ __forceinline__ float bf2f(__bf16 x) {
    unsigned short h = __builtin_bit_cast(unsigned short, x);
    union { unsigned u; float f; } v; v.u = ((unsigned)h) << 16;
    return v.f;
}
__device__ __forceinline__ float sigmoidf(float x) { return 1.0f / (1.0f + __expf(-x)); }

__device__ __forceinline__ void stv(float*  p, float v) { *p = v; }
__device__ __forceinline__ void stv(__bf16* p, float v) { *p = to_bf16(v); }

// Async global->LDS copy of 16 bytes per lane (CDNA5 TDM-adjacent async path).
__device__ __forceinline__ void async_cp_b128(unsigned lds_off, const void* gaddr) {
    unsigned long long ga = (unsigned long long)(uintptr_t)gaddr;
    asm volatile("global_load_async_to_lds_b128 %0, %1, off"
                 :: "v"(lds_off), "v"(ga) : "memory");
}
__device__ __forceinline__ void wait_async0() {
    asm volatile("s_wait_asynccnt 0x0" ::: "memory");
}
__device__ __forceinline__ void wait_async2() {
    asm volatile("s_wait_asynccnt 0x2" ::: "memory");
}

// ---------------------------------------------------------------------------
// Elementwise helpers
// ---------------------------------------------------------------------------
__global__ void cvt_f32_bf16_k(const float* __restrict__ in, __bf16* __restrict__ out, int n) {
    int i = blockIdx.x * 256 + threadIdx.x;
    if (i < n) out[i] = to_bf16(in[i]);
}

// out[N][K] = bf16(in[K][N])   (weights -> transposed bf16 for WMMA B-frags)
__global__ void transpose_f32_bf16_k(const float* __restrict__ in, __bf16* __restrict__ out,
                                     int K, int N) {
    int idx = blockIdx.x * 256 + threadIdx.x;
    if (idx < K * N) {
        int n = idx / K, k = idx - n * K;
        out[idx] = to_bf16(in[(size_t)k * N + n]);
    }
}

__global__ void add2_bf16_k(const __bf16* __restrict__ a, const __bf16* __restrict__ b,
                            __bf16* __restrict__ o, int n) {
    int i = blockIdx.x * 256 + threadIdx.x;
    if (i < n) o[i] = to_bf16(bf2f(a[i]) + bf2f(b[i]));
}

__global__ void softmax2_k(const float* __restrict__ sp, float* __restrict__ s) {
    if (threadIdx.x == 0 && blockIdx.x == 0) {
        float m  = fmaxf(sp[0], sp[1]);
        float e0 = __expf(sp[0] - m), e1 = __expf(sp[1] - m);
        float d  = e0 + e1;
        s[0] = e0 / d; s[1] = e1 / d;
    }
}

__global__ void combine_out_k(const __bf16* __restrict__ a, const __bf16* __restrict__ b,
                              const __bf16* __restrict__ c, const float* __restrict__ s,
                              __bf16* __restrict__ o, int n) {
    int i = blockIdx.x * 256 + threadIdx.x;
    if (i < n) o[i] = to_bf16(bf2f(a[i]) + s[0] * bf2f(b[i]) + s[1] * bf2f(c[i]));
}

__global__ void init_h_k(const float* __restrict__ h0, float* __restrict__ hF,
                         __bf16* __restrict__ hB) {
    int i = blockIdx.x * 256 + threadIdx.x;            // 64*512
    if (i < 64 * 512) { float v = h0[i & 511]; hF[i] = v; hB[i] = to_bf16(v); }
}

// ---------------------------------------------------------------------------
// Batched GEMM: C[M,N] = A[M,K](bf16) @ Bt[N,K]^T(bf16) + bias[N]
// block = 128 threads (4 waves); WG tile = 64x64; wave tile = 16(M) x 64(N).
// B k-slices (64x32) staged in LDS via async copies, double-buffered; all 4
// waves share the staged B fragments through ds_load_b128.
// ---------------------------------------------------------------------------
#define BSTRIDE 40            // 32 + 8 pad elements per staged B row (bank spread)
#define BBUF    (64 * BSTRIDE) // elements per staging buffer (5120B)

template <typename OutT>
__global__ void __launch_bounds__(128)
gemm_bf16_wmma(const __bf16* __restrict__ A, const __bf16* __restrict__ Bt,
               const float* __restrict__ bias, OutT* __restrict__ C,
               int K, int lda, int ldb, int ldc) {
    __shared__ __bf16 Bs[2 * BBUF];

    const int w    = threadIdx.x >> 5;
    const int lane = threadIdx.x & 31;
    const int ln   = lane & 15;
    const int kh   = lane >> 4;
    const int wgN  = blockIdx.x * 64;
    const int wgM  = blockIdx.y * 64;

    const unsigned ldsBase = (unsigned)(uintptr_t)&Bs[0];

    // Stage the 64(N) x 32(K) B slice at k0 into buffer `buf`.
    // 256 chunks of 16B; each of the 128 threads copies 2 chunks.
    auto stageB = [&](int k0, int buf) {
#pragma unroll
        for (int j = 0; j < 2; ++j) {
            int c  = threadIdx.x + j * 128;
            int n  = c >> 2;
            int ch = c & 3;
            const __bf16* gsrc = Bt + (size_t)(wgN + n) * ldb + k0 + ch * 8;
            unsigned dst = ldsBase + (unsigned)((buf * BBUF + n * BSTRIDE + ch * 8) * 2);
            async_cp_b128(dst, gsrc);
        }
    };

    v8f acc[4] = {};
    const __bf16* arow = A + (size_t)(wgM + w * 16 + ln) * lda;
    const int nk = K >> 5;

    stageB(0, 0);
    for (int i = 0; i < nk; ++i) {
        const int k0 = i << 5;
        if (i + 1 < nk) { stageB(k0 + 32, (i + 1) & 1); wait_async2(); }
        else            { wait_async0(); }
        __syncthreads();                       // staged slice i visible to all waves

        __builtin_prefetch((const void*)(arow + k0 + 64), 0, 1);  // global_prefetch_b8
        FragBF a;
        a.q[0] = *(const f32x4*)(arow + k0 + kh * 16);
        a.q[1] = *(const f32x4*)(arow + k0 + kh * 16 + 8);

        const __bf16* bbase = &Bs[(i & 1) * BBUF];
#pragma unroll
        for (int j = 0; j < 4; ++j) {
            const __bf16* bp = bbase + (j * 16 + ln) * BSTRIDE + kh * 16;
            FragBF b;
            b.q[0] = *(const f32x4*)(bp);
            b.q[1] = *(const f32x4*)(bp + 8);
            acc[j] = __builtin_amdgcn_wmma_f32_16x16x32_bf16(
                false, a.v, false, b.v, (short)0, acc[j], false, false);
        }
        __syncthreads();                       // readers done before buffer reuse
    }

#pragma unroll
    for (int j = 0; j < 4; ++j) {
        int n = wgN + j * 16 + ln;
        float bv = bias ? bias[n] : 0.0f;
#pragma unroll
        for (int i = 0; i < 8; ++i) {
            int r = wgM + w * 16 + kh * 8 + i;
            stv(C + (size_t)r * ldc + n, acc[j][i] + bv);
        }
    }
}

// ---------------------------------------------------------------------------
// Persistent GRU recurrence over T=1024 steps.
// grid = 32 WGs (one per 16-col H slice), block = 128 (4 waves, M-tiles of 16).
// Each WG keeps its U^T slice (3 gates x 16 cols x 512 K, bf16) in LDS,
// loaded once via async copies. h ping-pongs globally; grid barrier per step.
// ---------------------------------------------------------------------------
__global__ void __launch_bounds__(128)
gru_layer_persistent(const __bf16* __restrict__ xs,    // [B*T][1536] bf16 (x@W + bi)
                     const __bf16* __restrict__ Ut,    // [1536][512]  bf16 (U^T)
                     const float*  __restrict__ br,    // [1536]
                     float* hF0, float* hF1,           // fp32 h state (ping-pong)
                     __bf16* hB0, __bf16* hB1,         // bf16 h state (ping-pong)
                     __bf16* __restrict__ outSeq,      // [B*T][512] bf16 (pre-zoneout)
                     unsigned* __restrict__ cnt) {
    __shared__ __bf16 Us[48 * 520];                    // 520 = 512 + 8 pad (bank spread)

    const int w     = threadIdx.x >> 5;
    const int lane  = threadIdx.x & 31;
    const int ln    = lane & 15;
    const int kh    = lane >> 4;
    const int hbase = blockIdx.x * 16;
    const unsigned NWG = gridDim.x;

    // Async-stage this WG's U^T slice into LDS (rows: gate*16 + col).
    const unsigned usBase = (unsigned)(uintptr_t)&Us[0];
    for (int idx = threadIdx.x; idx < 48 * 64; idx += 128) {
        int rl = idx >> 6;
        int ck = (idx & 63) << 3;
        int g = rl >> 4, c = rl & 15;
        const __bf16* src = Ut + (size_t)(g * 512 + hbase + c) * 512 + ck;
        async_cp_b128(usBase + (unsigned)((rl * 520 + ck) * 2), src);
    }
    wait_async0();
    __syncthreads();

    const int hidx = hbase + ln;
    const float brz = br[hidx], brr = br[512 + hidx], brh = br[1024 + hidx];

    for (int t = 0; t < 1024; ++t) {
        const __bf16* hBc = (t & 1) ? hB1 : hB0;
        const float*  hFc = (t & 1) ? hF1 : hF0;
        __bf16*       hBn = (t & 1) ? hB0 : hB1;
        float*        hFn = (t & 1) ? hF0 : hF1;

        v8f az = {}, ar = {}, ah = {};
        const __bf16* hrow = hBc + (size_t)(w * 16 + ln) * 512;
#pragma unroll 4
        for (int k0 = 0; k0 < 512; k0 += 32) {
            FragBF a;
            a.q[0] = *(const f32x4*)(hrow + k0 + kh * 16);
            a.q[1] = *(const f32x4*)(hrow + k0 + kh * 16 + 8);
            FragBF b;
            const __bf16* l0 = &Us[(size_t)(ln) * 520 + k0 + kh * 16];
            b.q[0] = *(const f32x4*)l0; b.q[1] = *(const f32x4*)(l0 + 8);
            az = __builtin_amdgcn_wmma_f32_16x16x32_bf16(false, a.v, false, b.v, (short)0, az, false, false);
            const __bf16* l1 = &Us[(size_t)(16 + ln) * 520 + k0 + kh * 16];
            b.q[0] = *(const f32x4*)l1; b.q[1] = *(const f32x4*)(l1 + 8);
            ar = __builtin_amdgcn_wmma_f32_16x16x32_bf16(false, a.v, false, b.v, (short)0, ar, false, false);
            const __bf16* l2 = &Us[(size_t)(32 + ln) * 520 + k0 + kh * 16];
            b.q[0] = *(const f32x4*)l2; b.q[1] = *(const f32x4*)(l2 + 8);
            ah = __builtin_amdgcn_wmma_f32_16x16x32_bf16(false, a.v, false, b.v, (short)0, ah, false, false);
        }

        // Fused gate math + zoneout; each thread owns 8 (row,col) cells.
#pragma unroll
        for (int i = 0; i < 8; ++i) {
            int brow = w * 16 + kh * 8 + i;                    // batch row
            size_t xb = ((size_t)brow * 1024 + (size_t)t) * 1536;
            float xz = bf2f(xs[xb + hidx]);
            float xr = bf2f(xs[xb + 512 + hidx]);
            float xh = bf2f(xs[xb + 1024 + hidx]);
            float hp = hFc[brow * 512 + hidx];
            float z  = sigmoidf(xz + az[i] + brz);
            float r  = sigmoidf(xr + ar[i] + brr);
            float hh = tanhf(xh + r * (ah[i] + brh));
            float hn = z * hp + (1.0f - z) * hh;               // sequence output
            float hx = hn + 0.1f * (hp - hn);                  // zoneout carry
            outSeq[((size_t)brow * 1024 + (size_t)t) * 512 + hidx] = to_bf16(hn);
            hFn[brow * 512 + hidx] = hx;
            hBn[brow * 512 + hidx] = to_bf16(hx);
        }

        // Grid-wide step barrier (monotonic counter).
        __syncthreads();
        if (threadIdx.x == 0) {
            __hip_atomic_fetch_add(cnt, 1u, __ATOMIC_RELEASE, __HIP_MEMORY_SCOPE_AGENT);
            unsigned target = NWG * (unsigned)(t + 1);
            while (__hip_atomic_load(cnt, __ATOMIC_ACQUIRE, __HIP_MEMORY_SCOPE_AGENT) < target) {
                __builtin_amdgcn_s_sleep(2);
            }
        }
        __syncthreads();
    }
}

// ---------------------------------------------------------------------------
// Host-side orchestration
// ---------------------------------------------------------------------------
extern "C" void kernel_launch(void* const* d_in, const int* in_sizes, int n_in,
                              void* d_out, int out_size, void* d_ws, size_t ws_size,
                              hipStream_t stream) {
    const int B = 64, T = 1024, D = 256, H = 512;
    const int BT = B * T;                 // 65536
    const int G3 = 3 * H;                 // 1536

    const float* x    = (const float*)d_in[0];
    const float* h0_0 = (const float*)d_in[1];
    const float* h0_1 = (const float*)d_in[2];
    const float* h0_2 = (const float*)d_in[3];
    const float* W0   = (const float*)d_in[4];
    const float* U0   = (const float*)d_in[5];
    const float* bi0  = (const float*)d_in[6];
    const float* br0  = (const float*)d_in[7];
    const float* W1   = (const float*)d_in[8];
    const float* U1   = (const float*)d_in[9];
    const float* bi1  = (const float*)d_in[10];
    const float* br1  = (const float*)d_in[11];
    const float* W2   = (const float*)d_in[12];
    const float* U2   = (const float*)d_in[13];
    const float* bi2  = (const float*)d_in[14];
    const float* br2  = (const float*)d_in[15];
    const float* s_pre= (const float*)d_in[16];
    const float* Wd   = (const float*)d_in[17];
    const float* bd   = (const float*)d_in[18];
    float* outp = (float*)d_out;

    // Workspace carve-out (~480 MB total).
    char* p = (char*)d_ws;
    auto alloc = [&](size_t bytes) -> char* {
        char* r = p; p += (bytes + 255) & ~(size_t)255; return r;
    };
    unsigned* cnt = (unsigned*)alloc(256);
    float*    sws = (float*)((char*)cnt + 16);
    __bf16* W0t = (__bf16*)alloc((size_t)G3 * D * 2);
    __bf16* W1t = (__bf16*)alloc((size_t)G3 * H * 2);
    __bf16* W2t = (__bf16*)alloc((size_t)G3 * H * 2);
    __bf16* U0t = (__bf16*)alloc((size_t)G3 * H * 2);
    __bf16* U1t = (__bf16*)alloc((size_t)G3 * H * 2);
    __bf16* U2t = (__bf16*)alloc((size_t)G3 * H * 2);
    __bf16* Wdt = (__bf16*)alloc((size_t)D * H * 2);
    float*  hF0 = (float*) alloc((size_t)B * H * 4);
    float*  hF1 = (float*) alloc((size_t)B * H * 4);
    __bf16* hB0 = (__bf16*)alloc((size_t)B * H * 2);
    __bf16* hB1 = (__bf16*)alloc((size_t)B * H * 2);
    __bf16* Abuf  = (__bf16*)alloc((size_t)BT * H * 2);   // x(bf16)/x2/combined
    __bf16* xsb   = (__bf16*)alloc((size_t)BT * G3 * 2);  // input projections
    __bf16* out0  = (__bf16*)alloc((size_t)BT * H * 2);
    __bf16* pred1 = (__bf16*)alloc((size_t)BT * H * 2);
    __bf16* pred2 = (__bf16*)alloc((size_t)BT * H * 2);
    (void)ws_size; (void)in_sizes; (void)n_in; (void)out_size;

    auto blk = [](int n) { return dim3((n + 255) / 256); };

    // Weight conversion / transposition (cheap vs. total work; deterministic).
    transpose_f32_bf16_k<<<blk(D * G3), 256, 0, stream>>>(W0, W0t, D, G3);
    transpose_f32_bf16_k<<<blk(H * G3), 256, 0, stream>>>(W1, W1t, H, G3);
    transpose_f32_bf16_k<<<blk(H * G3), 256, 0, stream>>>(W2, W2t, H, G3);
    transpose_f32_bf16_k<<<blk(H * G3), 256, 0, stream>>>(U0, U0t, H, G3);
    transpose_f32_bf16_k<<<blk(H * G3), 256, 0, stream>>>(U1, U1t, H, G3);
    transpose_f32_bf16_k<<<blk(H * G3), 256, 0, stream>>>(U2, U2t, H, G3);
    transpose_f32_bf16_k<<<blk(H * D),  256, 0, stream>>>(Wd, Wdt, H, D);

    // ---------------- layer 0 ----------------
    cvt_f32_bf16_k<<<blk(BT * D), 256, 0, stream>>>(x, Abuf, BT * D);
    gemm_bf16_wmma<__bf16><<<dim3(G3 / 64, BT / 64), 128, 0, stream>>>(
        Abuf, W0t, bi0, xsb, D, D, D, G3);
    init_h_k<<<blk(B * H), 256, 0, stream>>>(h0_0, hF0, hB0);
    hipMemsetAsync(cnt, 0, sizeof(unsigned), stream);
    gru_layer_persistent<<<dim3(H / 16), 128, 0, stream>>>(
        xsb, U0t, br0, hF0, hF1, hB0, hB1, out0, cnt);

    // ---------------- layer 1 ----------------
    gemm_bf16_wmma<__bf16><<<dim3(G3 / 64, BT / 64), 128, 0, stream>>>(
        out0, W1t, bi1, xsb, H, H, H, G3);
    init_h_k<<<blk(B * H), 256, 0, stream>>>(h0_1, hF0, hB0);
    hipMemsetAsync(cnt, 0, sizeof(unsigned), stream);
    gru_layer_persistent<<<dim3(H / 16), 128, 0, stream>>>(
        xsb, U1t, br1, hF0, hF1, hB0, hB1, pred1, cnt);

    // ---------------- layer 2 (skip in) ----------------
    add2_bf16_k<<<blk(BT * H), 256, 0, stream>>>(out0, pred1, Abuf, BT * H);
    gemm_bf16_wmma<__bf16><<<dim3(G3 / 64, BT / 64), 128, 0, stream>>>(
        Abuf, W2t, bi2, xsb, H, H, H, G3);
    init_h_k<<<blk(B * H), 256, 0, stream>>>(h0_2, hF0, hB0);
    hipMemsetAsync(cnt, 0, sizeof(unsigned), stream);
    gru_layer_persistent<<<dim3(H / 16), 128, 0, stream>>>(
        xsb, U2t, br2, hF0, hF1, hB0, hB1, pred2, cnt);

    // ---------------- skip combine + dense ----------------
    softmax2_k<<<1, 32, 0, stream>>>(s_pre, sws);
    combine_out_k<<<blk(BT * H), 256, 0, stream>>>(out0, pred1, pred2, sws, Abuf, BT * H);
    gemm_bf16_wmma<float><<<dim3(D / 64, BT / 64), 128, 0, stream>>>(
        Abuf, Wdt, bd, outp, H, H, H, D);
}
